// SparseAttention_81097572483515
// MI455X (gfx1250) — matrix operand order repaired and checked by reading, not compiled, not run on hardware
//
#include <hip/hip_runtime.h>
#include <cstdint>
#include <cstddef>

typedef _Float16 half_t;
typedef __attribute__((ext_vector_type(16))) _Float16 v16h;
typedef __attribute__((ext_vector_type(8)))  float    v8f;

#define S_LEN 2048
#define DIM_  1024
#define HEADS_ 16
#define KVH_  2
#define GQ_   8
#define DH_   64
#define NCB_  32
#define NSEL_ 8
#define WIN_  512
#define CDIM_ 4096
#define QKVN_ 1280
#define GATESN_ 64   /* 48 padded to 64 */
#define SCALE_ 0.125f
#define MASKNEG_ (-3.0e38f)
#define MINIT_   (-1.0e38f)

// ---------------- WMMA fragment helpers (gfx1250 wave32 layouts) -------------

__device__ inline v16h zero16h() {
  v16h z;
#pragma unroll
  for (int j = 0; j < 16; ++j) z[j] = (half_t)0.f;
  return z;
}

// A-matrix 16x32 f16: lanes 0-15 -> M=lane, K {0..7,16..23}; lanes 16-31 -> K {8..15,24..31}
// src must be 16B aligned, ld (halves) multiple of 8.
__device__ inline v16h load_a_frag(const half_t* src, int ld, int lane) {
  int m = lane & 15;
  int kb = (lane < 16) ? 0 : 8;
  const half_t* pp = src + (size_t)m * ld + kb;
  union { uint4 u[2]; v16h v; } t;
  t.u[0] = *(const uint4*)(pp);
  t.u[1] = *(const uint4*)(pp + 16);
  return t.v;
}

// B-matrix 32x16 f16 loaded from a column-major view srcT[n][k]:
// lanes 0-15 -> N=lane, K {0..7,16..23} ; lanes 16-31 -> K {8..15,24..31}
__device__ inline v16h load_b_frag(const half_t* srcT, int ld, int lane) {
  int n = lane & 15;
  int ks = (lane < 16) ? 0 : 8;
  const half_t* pp = srcT + (size_t)n * ld + ks;
  union { uint4 u[2]; v16h v; } t;
  t.u[0] = *(const uint4*)(pp);
  t.u[1] = *(const uint4*)(pp + 16);
  return t.v;
}

__device__ inline v8f wmma_f16(v16h a, v16h b, v8f c) {
  return __builtin_amdgcn_wmma_f32_16x16x32_f16(false, a, false, b, (short)0, c,
                                                false, false);
}

// reductions across the 16-lane column groups (rows of the D fragment)
__device__ inline float rmax16(float v) {
  v = fmaxf(v, __shfl_xor(v, 1));
  v = fmaxf(v, __shfl_xor(v, 2));
  v = fmaxf(v, __shfl_xor(v, 4));
  v = fmaxf(v, __shfl_xor(v, 8));
  return v;
}
__device__ inline float rsum16(float v) {
  v += __shfl_xor(v, 1); v += __shfl_xor(v, 2);
  v += __shfl_xor(v, 4); v += __shfl_xor(v, 8);
  return v;
}

// ---------------- elementwise kernels ---------------------------------------

__global__ void rmsnorm_kernel(const float* __restrict__ x, const float* __restrict__ g,
                               float* __restrict__ xn, half_t* __restrict__ xnh) {
  int row = blockIdx.x;
  const float* xr = x + (size_t)row * DIM_;
  __shared__ float red[256];
  float s = 0.f;
  for (int c = threadIdx.x; c < DIM_; c += 256) { float v = xr[c]; s += v * v; }
  red[threadIdx.x] = s;
  __syncthreads();
  for (int off = 128; off > 0; off >>= 1) {
    if (threadIdx.x < off) red[threadIdx.x] += red[threadIdx.x + off];
    __syncthreads();
  }
  float r = rsqrtf(red[0] / (float)DIM_ + 1.1920929e-07f);
  for (int c = threadIdx.x; c < DIM_; c += 256) {
    float v = xr[c] * r * g[c];
    xn[(size_t)row * DIM_ + c] = v;
    xnh[(size_t)row * DIM_ + c] = (half_t)v;
  }
}

__global__ void cvt_kernel(const float* __restrict__ in, half_t* __restrict__ out, size_t n) {
  size_t i = (size_t)blockIdx.x * 256 + threadIdx.x;
  if (i < n) out[i] = (half_t)in[i];
}

// zero-padded f32->f16 matrix convert (pads columns incols..outcols-1 with 0)
__global__ void padcvt_kernel(const float* __restrict__ in, half_t* __restrict__ out,
                              int rows, int incols, int outcols) {
  int idx = blockIdx.x * 256 + threadIdx.x;
  if (idx >= rows * outcols) return;
  int r = idx / outcols, c = idx % outcols;
  out[idx] = (c < incols) ? (half_t)in[(size_t)r * incols + c] : (half_t)0.f;
}

__global__ void padbias_kernel(const float* __restrict__ in, float* __restrict__ out,
                               int incols) {
  int c = threadIdx.x;
  out[c] = (c < incols) ? in[c] : 0.f;
}

// ---------------- generic WMMA GEMM -----------------------------------------
// C[M,N] = act(A[M,K] @ B[K,N] + bias). A,B f16 row-major. M%64==0, N%64==0, K%32==0.
// act: 0 none, 1 relu, 2 sigmoid.  outmode: 0 -> f32, 1 -> f16.
#define GBM 64
#define GBN 64
#define GBK 32
#define BPAD 8  /* keeps sBt rows 16B aligned */
__global__ __launch_bounds__(256) void gemm_wmma_kernel(
    const half_t* __restrict__ A, const half_t* __restrict__ B,
    const float* __restrict__ bias, float* __restrict__ Cf32,
    half_t* __restrict__ Cf16, int M, int N, int K, int act, int outmode) {
  __shared__ half_t sA[GBM][GBK];
  __shared__ half_t sBt[GBN][GBK + BPAD];
  int bm = blockIdx.x * GBM;
  int bn = blockIdx.y * GBN;
  int tid = threadIdx.x;
  int lane = tid & 31;
  int wave = tid >> 5;
  int wm = wave & 3;
  int wn = wave >> 2;
  v8f acc0 = {}, acc1 = {};

  int arow = tid >> 2, aseg = (tid & 3) * 8;
  int bk = tid >> 3, bnseg = (tid & 7) * 8;
  const half_t* aptr = A + (size_t)(bm + arow) * K + aseg;
  const half_t* bptr = B + (size_t)bk * N + bn + bnseg;

  for (int k0 = 0; k0 < K; k0 += GBK) {
    // A tile 64x32: one b128 load + one b128 LDS store per thread
    *(uint4*)&sA[arow][aseg] = *(const uint4*)(aptr + k0);
    // B tile 32x64, transposed into LDS
    {
      uint4 bv = *(const uint4*)(bptr + (size_t)k0 * N);
      const half_t* hh = (const half_t*)&bv;
#pragma unroll
      for (int j = 0; j < 8; ++j) sBt[bnseg + j][bk] = hh[j];
    }
    if (k0 + GBK < K) {  // prefetch next tiles (global_prefetch_b8)
      __builtin_prefetch(aptr + k0 + GBK, 0, 1);
      __builtin_prefetch(bptr + (size_t)(k0 + GBK) * N, 0, 1);
    }
    __syncthreads();
    v16h a = load_a_frag(&sA[wm * 16][0], GBK, lane);
    v16h b0 = load_b_frag(&sBt[wn * 32][0], GBK + BPAD, lane);
    v16h b1 = load_b_frag(&sBt[wn * 32 + 16][0], GBK + BPAD, lane);
    acc0 = wmma_f16(a, b0, acc0);
    acc1 = wmma_f16(a, b1, acc1);
    __syncthreads();
  }

  int col = lane & 15;
  int rbase = (lane < 16) ? 0 : 8;
#pragma unroll
  for (int r = 0; r < 8; ++r) {
    int row = bm + wm * 16 + rbase + r;
#pragma unroll
    for (int t = 0; t < 2; ++t) {
      float v = (t == 0) ? acc0[r] : acc1[r];
      int c = bn + wn * 32 + t * 16 + col;
      if (bias) v += bias[c];
      if (act == 1) v = fmaxf(v, 0.f);
      else if (act == 2) v = 1.f / (1.f + __expf(-v));
      if (outmode == 0) Cf32[(size_t)row * N + c] = v;
      else Cf16[(size_t)row * N + c] = (half_t)v;
    }
  }
}

// ---------------- RoPE / reshape / MLP-input prep ---------------------------
__global__ void prep_kernel(const float* __restrict__ qkv,
                            const float* __restrict__ k_pos, const float* __restrict__ v_pos,
                            half_t* __restrict__ q_h, half_t* __restrict__ rq_h,
                            half_t* __restrict__ rk_h, half_t* __restrict__ v_h,
                            float* __restrict__ qbar,
                            half_t* __restrict__ kcm, half_t* __restrict__ vcm) {
  int s = blockIdx.x;
  int d = threadIdx.x;  // 0..63
  float inv = __powf(10000.f, -(float)(d & ~1) / (float)DH_);
  float ang = (float)s * inv;
  float cs = __cosf(ang), sn = __sinf(ang);
  const float* row = qkv + (size_t)s * QKVN_;

  for (int h = 0; h < HEADS_; ++h) {
    float qd = row[h * DH_ + d];
    float qp = row[h * DH_ + (d ^ 1)];
    float rot = (d & 1) ? qp : -qp;
    float rq = qd * cs + rot * sn;
    size_t o = ((size_t)h * S_LEN + s) * DH_ + d;
    q_h[o] = (half_t)qd;
    rq_h[o] = (half_t)rq;
  }
  int cb = s >> 6, sb = s & 63;
  for (int kvh = 0; kvh < KVH_; ++kvh) {
    float kd = row[HEADS_ * DH_ + kvh * DH_ + d];
    float kp = row[HEADS_ * DH_ + kvh * DH_ + (d ^ 1)];
    float rot = (d & 1) ? kp : -kp;
    float rk = kd * cs + rot * sn;
    float vd = row[(HEADS_ + KVH_) * DH_ + kvh * DH_ + d];
    size_t o = ((size_t)kvh * S_LEN + s) * DH_ + d;
    rk_h[o] = (half_t)rk;
    v_h[o] = (half_t)vd;
    kcm[((size_t)(kvh * NCB_ + cb)) * CDIM_ + sb * DH_ + d] =
        (half_t)(kd + k_pos[((size_t)kvh * 64 + sb) * DH_ + d]);
    vcm[((size_t)(kvh * NCB_ + cb)) * CDIM_ + sb * DH_ + d] =
        (half_t)(vd + v_pos[((size_t)kvh * 64 + sb) * DH_ + d]);
    float acc = 0.f;
    for (int g = 0; g < GQ_; ++g) acc += row[(kvh * GQ_ + g) * DH_ + d];
    qbar[o] = acc * 0.125f;
  }
}

// ---------------- build padded compressed K/V (mem + 32 blocks + zero pad) ---
__global__ void build_ckcv_kernel(const float* __restrict__ ckb, const float* __restrict__ cvb,
                                  const float* __restrict__ mem_kv,
                                  half_t* __restrict__ ckh, half_t* __restrict__ cvh) {
  int j = blockIdx.x;   // 0..47
  int kvh = blockIdx.y; // 0..1
  int d = threadIdx.x;  // 0..63
  float kv = 0.f, vv = 0.f;
  if (j == 0) {
    kv = mem_kv[((size_t)(0 * KVH_ + kvh)) * DH_ + d];
    vv = mem_kv[((size_t)(1 * KVH_ + kvh)) * DH_ + d];
  } else if (j <= NCB_) {
    int r = kvh * NCB_ + (j - 1);
    kv = ckb[(size_t)r * DH_ + d];
    vv = cvb[(size_t)r * DH_ + d];
  }
  ckh[((size_t)kvh * 48 + j) * DH_ + d] = (half_t)kv;
  cvh[((size_t)kvh * 48 + j) * DH_ + d] = (half_t)vv;
}

// ---------------- compressed attention (33 keys padded to 48) ----------------
__global__ __launch_bounds__(32) void cattn_kernel(const half_t* __restrict__ q_h,
                                                   const half_t* __restrict__ ckh,
                                                   const half_t* __restrict__ cvh,
                                                   float* __restrict__ c_out) {
  int qt = blockIdx.x, h = blockIdx.y, kvh = h >> 3;
  int lane = threadIdx.x;
  __shared__ half_t sP[16][64];
  __shared__ half_t sVT[64][64];  // [d][key], keys 48..63 zero
  for (int idx = lane; idx < 64 * 64; idx += 32) {
    int d = idx >> 6, k = idx & 63;
    sVT[d][k] = (k < 48) ? cvh[((size_t)kvh * 48 + k) * DH_ + d] : (half_t)0.f;
  }
  for (int idx = lane; idx < 16 * 16; idx += 32) sP[idx >> 4][48 + (idx & 15)] = (half_t)0.f;
  __syncthreads();

  const half_t* qp = q_h + ((size_t)h * S_LEN + qt * 16) * DH_;
  v16h a0 = load_a_frag(qp, DH_, lane);
  v16h a1 = load_a_frag(qp + 32, DH_, lane);

  int col = lane & 15, rb = (lane < 16) ? 0 : 8;
  v8f sc[3];
#pragma unroll
  for (int kt = 0; kt < 3; ++kt) {
    v8f acc = {};
    const half_t* kpp = ckh + ((size_t)kvh * 48 + kt * 16) * DH_;
    acc = wmma_f16(a0, load_b_frag(kpp, DH_, lane), acc);
    acc = wmma_f16(a1, load_b_frag(kpp + 32, DH_, lane), acc);
    int j = kt * 16 + col;
    int seq = (j == 0) ? -1 : (j <= NCB_ ? j * 64 - 1 : (1 << 30));
#pragma unroll
    for (int r = 0; r < 8; ++r) {
      int qpos = qt * 16 + rb + r;
      float v = acc[r] * SCALE_;
      acc[r] = (seq < qpos) ? v : MASKNEG_;
    }
    sc[kt] = acc;
  }
  // full softmax over 48 columns
#pragma unroll
  for (int r = 0; r < 8; ++r) {
    float mx = fmaxf(fmaxf(sc[0][r], sc[1][r]), sc[2][r]);
    mx = rmax16(mx);
    float e0 = __expf(sc[0][r] - mx);
    float e1 = __expf(sc[1][r] - mx);
    float e2 = __expf(sc[2][r] - mx);
    float sum = rsum16(e0 + e1 + e2);
    float invs = 1.f / sum;
    sP[rb + r][col] = (half_t)(e0 * invs);
    sP[rb + r][16 + col] = (half_t)(e1 * invs);
    sP[rb + r][32 + col] = (half_t)(e2 * invs);
  }
  __syncthreads();
  v16h pa0 = load_a_frag(&sP[0][0], 64, lane);
  v16h pa1 = load_a_frag(&sP[0][32], 64, lane);
#pragma unroll
  for (int nt = 0; nt < 4; ++nt) {
    v8f acc = {};
    acc = wmma_f16(pa0, load_b_frag(&sVT[nt * 16][0], 64, lane), acc);
    acc = wmma_f16(pa1, load_b_frag(&sVT[nt * 16][32], 64, lane), acc);
#pragma unroll
    for (int r = 0; r < 8; ++r) {
      int qpos = qt * 16 + rb + r;
      c_out[((size_t)h * S_LEN + qpos) * DH_ + nt * 16 + col] = acc[r];
    }
  }
}

// ---------------- block selection (top-8 of masked softmaxed importance) -----
__global__ void select_kernel(const float* __restrict__ qbar, const float* __restrict__ ckb,
                              int* __restrict__ sel_idx, int* __restrict__ sel_ok) {
  int i = blockIdx.x * 64 + threadIdx.x;
  int kvh = blockIdx.y;
  if (i >= S_LEN) return;
  const float* qb = qbar + ((size_t)kvh * S_LEN + i) * DH_;
  float qr[DH_];
#pragma unroll
  for (int d = 0; d < DH_; ++d) qr[d] = qb[d];
  float imp[NCB_];
  float mx = -1000.f;  // sentinel participates in softmax
  for (int w = 0; w < NCB_; ++w) {
    if ((w + 1) * 64 - 1 < i) {
      const float* cr = ckb + ((size_t)(kvh * NCB_ + w)) * DH_;
      float s = 0.f;
      for (int d = 0; d < DH_; ++d) s += qr[d] * cr[d];
      s *= SCALE_;
      imp[w] = s;
      mx = fmaxf(mx, s);
    } else imp[w] = MASKNEG_;
  }
  float denom = __expf(-1000.f - mx);
  float val[NCB_];
  for (int w = 0; w < NCB_; ++w) { val[w] = __expf(imp[w] - mx); denom += val[w]; }
  float invd = 1.f / denom;
  for (int w = 0; w < NCB_; ++w) val[w] *= invd;
  bool used[NCB_] = {false};
  for (int t = 0; t < NSEL_; ++t) {
    int best = 0; float bv = -1.f;
    for (int w = 0; w < NCB_; ++w)
      if (!used[w] && val[w] > bv) { bv = val[w]; best = w; }
    used[best] = true;
    sel_idx[((size_t)kvh * S_LEN + i) * NSEL_ + t] = best;
    sel_ok[((size_t)kvh * S_LEN + i) * NSEL_ + t] = (bv > 1e-10f) ? 1 : 0;
  }
}

// ---------------- fine attention over gathered blocks ------------------------
__global__ __launch_bounds__(32) void fattn_kernel(const half_t* __restrict__ rq_h,
                                                   const half_t* __restrict__ rk_h,
                                                   const half_t* __restrict__ v_h,
                                                   const int* __restrict__ sel_idx,
                                                   const int* __restrict__ sel_ok,
                                                   float* __restrict__ f_out) {
  int i = blockIdx.x, kvh = blockIdx.y, lane = threadIdx.x;
  __shared__ half_t sP[16][32];
  __shared__ half_t sVT[64][32];
  int blk[9], ok[9];
#pragma unroll
  for (int t = 0; t < 8; ++t) {
    blk[t] = sel_idx[((size_t)kvh * S_LEN + i) * NSEL_ + t];
    ok[t] = sel_ok[((size_t)kvh * S_LEN + i) * NSEL_ + t];
  }
  blk[8] = i >> 6; ok[8] = 1;
  int m_ = lane & 15, kb = (lane < 16) ? 0 : 8, rb = (lane < 16) ? 0 : 8, col = lane & 15;
  v16h a0 = zero16h(), a1 = zero16h();
  if (m_ < GQ_) {  // 8 real rows = grouped heads, rest zero padding
    const half_t* p = rq_h + ((size_t)(kvh * GQ_ + m_) * S_LEN + i) * DH_;
    union { uint4 u[2]; v16h v; } t0, t1;
    t0.u[0] = *(const uint4*)(p + kb);
    t0.u[1] = *(const uint4*)(p + 16 + kb);
    t1.u[0] = *(const uint4*)(p + 32 + kb);
    t1.u[1] = *(const uint4*)(p + 48 + kb);
    a0 = t0.v; a1 = t1.v;
  }
  float rm[8], rl[8];
  v8f o0 = {}, o1 = {}, o2 = {}, o3 = {};
#pragma unroll
  for (int r = 0; r < 8; ++r) { rm[r] = MINIT_; rl[r] = 0.f; }

  for (int b = 0; b < 9; ++b) {
    if (!ok[b]) continue;
    for (int hf = 0; hf < 2; ++hf) {
      int kb0 = blk[b] * 64 + hf * 32;
      if (b == 8 && kb0 > i) continue;  // fully non-causal sub-tile of diag block
      const half_t* kpp = rk_h + ((size_t)kvh * S_LEN + kb0) * DH_;
      v8f s0 = {}, s1 = {};
      s0 = wmma_f16(a0, load_b_frag(kpp, DH_, lane), s0);
      s0 = wmma_f16(a1, load_b_frag(kpp + 32, DH_, lane), s0);
      s1 = wmma_f16(a0, load_b_frag(kpp + 16 * DH_, DH_, lane), s1);
      s1 = wmma_f16(a1, load_b_frag(kpp + 16 * DH_ + 32, DH_, lane), s1);
      bool v0 = true, v1 = true;
      if (b == 8) { v0 = (kb0 + col) <= i; v1 = (kb0 + 16 + col) <= i; }
      float al[8];
#pragma unroll
      for (int r = 0; r < 8; ++r) {
        s0[r] = v0 ? s0[r] * SCALE_ : MASKNEG_;
        s1[r] = v1 ? s1[r] * SCALE_ : MASKNEG_;
        float mxn = rmax16(fmaxf(s0[r], s1[r]));
        float nm = fmaxf(rm[r], mxn);
        al[r] = __expf(rm[r] - nm);
        rm[r] = nm;
        float e0 = __expf(s0[r] - nm), e1 = __expf(s1[r] - nm);
        rl[r] = rl[r] * al[r] + rsum16(e0 + e1);
        sP[rb + r][col] = (half_t)e0;
        sP[rb + r][16 + col] = (half_t)e1;
      }
#pragma unroll
      for (int r = 0; r < 8; ++r) { o0[r] *= al[r]; o1[r] *= al[r]; o2[r] *= al[r]; o3[r] *= al[r]; }
      for (int idx = lane; idx < 64 * 32; idx += 32) {
        int d = idx >> 5, k = idx & 31;
        sVT[d][k] = v_h[((size_t)kvh * S_LEN + kb0 + k) * DH_ + d];
      }
      __syncthreads();
      v16h pa = load_a_frag(&sP[0][0], 32, lane);
      o0 = wmma_f16(pa, load_b_frag(&sVT[0][0], 32, lane), o0);
      o1 = wmma_f16(pa, load_b_frag(&sVT[16][0], 32, lane), o1);
      o2 = wmma_f16(pa, load_b_frag(&sVT[32][0], 32, lane), o2);
      o3 = wmma_f16(pa, load_b_frag(&sVT[48][0], 32, lane), o3);
      __syncthreads();
    }
  }
  if (lane < 16) {  // only rows 0..7 are real heads
#pragma unroll
    for (int r = 0; r < 8; ++r) {
      float inv = 1.f / rl[r];
      size_t base = ((size_t)(kvh * GQ_ + r) * S_LEN + i) * DH_;
      f_out[base + col] = o0[r] * inv;
      f_out[base + 16 + col] = o1[r] * inv;
      f_out[base + 32 + col] = o2[r] * inv;
      f_out[base + 48 + col] = o3[r] * inv;
    }
  }
}

// ---------------- exact causal sliding-window attention ----------------------
__global__ __launch_bounds__(32) void swin_kernel(const half_t* __restrict__ rq_h,
                                                  const half_t* __restrict__ rk_h,
                                                  const half_t* __restrict__ v_h,
                                                  float* __restrict__ s_out) {
  int qt = blockIdx.x, h = blockIdx.y, kvh = h >> 3, lane = threadIdx.x;
  int q0 = qt * 16;
  __shared__ half_t sP[16][32];
  __shared__ half_t sVT[64][32];
  int m_ = lane & 15, kb = (lane < 16) ? 0 : 8, rb = (lane < 16) ? 0 : 8, col = lane & 15;
  const half_t* qp = rq_h + ((size_t)h * S_LEN + q0) * DH_;
  v16h a0 = load_a_frag(qp, DH_, lane);
  v16h a1 = load_a_frag(qp + 32, DH_, lane);
  (void)m_; (void)kb;
  float rm[8], rl[8];
  v8f o0 = {}, o1 = {}, o2 = {}, o3 = {};
#pragma unroll
  for (int r = 0; r < 8; ++r) { rm[r] = MINIT_; rl[r] = 0.f; }

  int kstart = (q0 >= WIN_) ? ((q0 - WIN_) & ~31) : 0;
  for (int k0 = kstart; k0 <= q0 + 15; k0 += 32) {
    const half_t* kpp = rk_h + ((size_t)kvh * S_LEN + k0) * DH_;
    v8f s0 = {}, s1 = {};
    s0 = wmma_f16(a0, load_b_frag(kpp, DH_, lane), s0);
    s0 = wmma_f16(a1, load_b_frag(kpp + 32, DH_, lane), s0);
    s1 = wmma_f16(a0, load_b_frag(kpp + 16 * DH_, DH_, lane), s1);
    s1 = wmma_f16(a1, load_b_frag(kpp + 16 * DH_ + 32, DH_, lane), s1);
    float al[8];
    int k0c = k0 + col, k1c = k0 + 16 + col;
#pragma unroll
    for (int r = 0; r < 8; ++r) {
      int q = q0 + rb + r;
      bool v0 = (k0c <= q) && (q - k0c <= WIN_);
      bool v1 = (k1c <= q) && (q - k1c <= WIN_);
      s0[r] = v0 ? s0[r] * SCALE_ : MASKNEG_;
      s1[r] = v1 ? s1[r] * SCALE_ : MASKNEG_;
      float mxn = rmax16(fmaxf(s0[r], s1[r]));
      float nm = fmaxf(rm[r], mxn);
      al[r] = __expf(rm[r] - nm);
      rm[r] = nm;
      float e0 = __expf(s0[r] - nm), e1 = __expf(s1[r] - nm);
      rl[r] = rl[r] * al[r] + rsum16(e0 + e1);
      sP[rb + r][col] = (half_t)e0;
      sP[rb + r][16 + col] = (half_t)e1;
    }
#pragma unroll
    for (int r = 0; r < 8; ++r) { o0[r] *= al[r]; o1[r] *= al[r]; o2[r] *= al[r]; o3[r] *= al[r]; }
    for (int idx = lane; idx < 64 * 32; idx += 32) {
      int d = idx >> 5, k = idx & 31;
      sVT[d][k] = v_h[((size_t)kvh * S_LEN + k0 + k) * DH_ + d];
    }
    __syncthreads();
    v16h pa = load_a_frag(&sP[0][0], 32, lane);
    o0 = wmma_f16(pa, load_b_frag(&sVT[0][0], 32, lane), o0);
    o1 = wmma_f16(pa, load_b_frag(&sVT[16][0], 32, lane), o1);
    o2 = wmma_f16(pa, load_b_frag(&sVT[32][0], 32, lane), o2);
    o3 = wmma_f16(pa, load_b_frag(&sVT[48][0], 32, lane), o3);
    __syncthreads();
  }
#pragma unroll
  for (int r = 0; r < 8; ++r) {
    float inv = 1.f / rl[r];
    int q = q0 + rb + r;
    size_t base = ((size_t)h * S_LEN + q) * DH_;
    s_out[base + col] = o0[r] * inv;
    s_out[base + 16 + col] = o1[r] * inv;
    s_out[base + 32 + col] = o2[r] * inv;
    s_out[base + 48 + col] = o3[r] * inv;
  }
}

// ---------------- gated combine ----------------------------------------------
__global__ void combine_kernel(const float* __restrict__ gates, const float* __restrict__ c_out,
                               const float* __restrict__ f_out, const float* __restrict__ s_out,
                               half_t* __restrict__ comb) {
  int idx = blockIdx.x * 256 + threadIdx.x;  // over 2048*1024
  int s = idx >> 10, c = idx & 1023, h = c >> 6, d = c & 63;
  float g0 = gates[(size_t)s * GATESN_ + h * 3 + 0];
  float g1 = gates[(size_t)s * GATESN_ + h * 3 + 1];
  float g2 = gates[(size_t)s * GATESN_ + h * 3 + 2];
  size_t o = ((size_t)h * S_LEN + s) * DH_ + d;
  comb[idx] = (half_t)(g0 * c_out[o] + g1 * f_out[o] + g2 * s_out[o]);
}

// ---------------- host orchestration -----------------------------------------
extern "C" void kernel_launch(void* const* d_in, const int* in_sizes, int n_in,
                              void* d_out, int out_size, void* d_ws, size_t ws_size,
                              hipStream_t stream) {
  (void)in_sizes; (void)n_in; (void)out_size; (void)ws_size;
  const float* x      = (const float*)d_in[0];
  const float* rms_g  = (const float*)d_in[1];
  const float* W_qkv  = (const float*)d_in[2];
  const float* k_pos  = (const float*)d_in[3];
  const float* v_pos  = (const float*)d_in[4];
  const float* mem_kv = (const float*)d_in[5];
  const float* kW1    = (const float*)d_in[6];
  const float* kb1    = (const float*)d_in[7];
  const float* kW2    = (const float*)d_in[8];
  const float* kb2    = (const float*)d_in[9];
  const float* vW1    = (const float*)d_in[10];
  const float* vb1    = (const float*)d_in[11];
  const float* vW2    = (const float*)d_in[12];
  const float* vb2    = (const float*)d_in[13];
  const float* W_comb = (const float*)d_in[14];
  const float* b_comb = (const float*)d_in[15];
  const float* W_out  = (const float*)d_in[16];
  float* out = (float*)d_out;

  char* p = (char*)d_ws;
  auto alloc = [&](size_t bytes) -> char* {
    char* r = p;
    p += (bytes + 255) & ~(size_t)255;
    return r;
  };
  float*  xn_f32   = (float*) alloc((size_t)S_LEN * DIM_ * 4);
  half_t* xn_f16   = (half_t*)alloc((size_t)S_LEN * DIM_ * 2);
  half_t* Wqkv_h   = (half_t*)alloc((size_t)DIM_ * QKVN_ * 2);
  float*  qkv_f32  = (float*) alloc((size_t)S_LEN * QKVN_ * 4);
  half_t* q_h      = (half_t*)alloc((size_t)HEADS_ * S_LEN * DH_ * 2);
  half_t* rq_h     = (half_t*)alloc((size_t)HEADS_ * S_LEN * DH_ * 2);
  half_t* rk_h     = (half_t*)alloc((size_t)KVH_ * S_LEN * DH_ * 2);
  half_t* v_h      = (half_t*)alloc((size_t)KVH_ * S_LEN * DH_ * 2);
  float*  qbar     = (float*) alloc((size_t)KVH_ * S_LEN * DH_ * 4);
  half_t* kcm      = (half_t*)alloc((size_t)64 * CDIM_ * 2);
  half_t* vcm      = (half_t*)alloc((size_t)64 * CDIM_ * 2);
  half_t* kW1_h    = (half_t*)alloc((size_t)CDIM_ * CDIM_ * 2);
  half_t* vW1_h    = (half_t*)alloc((size_t)CDIM_ * CDIM_ * 2);
  half_t* kW2_h    = (half_t*)alloc((size_t)CDIM_ * DH_ * 2);
  half_t* vW2_h    = (half_t*)alloc((size_t)CDIM_ * DH_ * 2);
  half_t* hk_h     = (half_t*)alloc((size_t)64 * CDIM_ * 2);
  half_t* hv_h     = (half_t*)alloc((size_t)64 * CDIM_ * 2);
  float*  ckb_f32  = (float*) alloc((size_t)64 * DH_ * 4);
  float*  cvb_f32  = (float*) alloc((size_t)64 * DH_ * 4);
  half_t* ck_h     = (half_t*)alloc((size_t)KVH_ * 48 * DH_ * 2);
  half_t* cv_h     = (half_t*)alloc((size_t)KVH_ * 48 * DH_ * 2);
  int*    sel_idx  = (int*)   alloc((size_t)KVH_ * S_LEN * NSEL_ * 4);
  int*    sel_ok   = (int*)   alloc((size_t)KVH_ * S_LEN * NSEL_ * 4);
  float*  c_out    = (float*) alloc((size_t)HEADS_ * S_LEN * DH_ * 4);
  float*  f_out    = (float*) alloc((size_t)HEADS_ * S_LEN * DH_ * 4);
  float*  s_out    = (float*) alloc((size_t)HEADS_ * S_LEN * DH_ * 4);
  half_t* Wcomb_h  = (half_t*)alloc((size_t)DIM_ * GATESN_ * 2);
  float*  bcomb_p  = (float*) alloc((size_t)GATESN_ * 4);
  float*  gates    = (float*) alloc((size_t)S_LEN * GATESN_ * 4);
  half_t* comb_h   = (half_t*)alloc((size_t)S_LEN * DIM_ * 2);
  half_t* Wout_h   = (half_t*)alloc((size_t)DIM_ * DIM_ * 2);

  auto cvt = [&](const float* src, half_t* dst, size_t n) {
    cvt_kernel<<<dim3((unsigned)((n + 255) / 256)), 256, 0, stream>>>(src, dst, n);
  };

  // 1. RMSNorm (+f16 copy)
  rmsnorm_kernel<<<S_LEN, 256, 0, stream>>>(x, rms_g, xn_f32, xn_f16);

  // 2. weight conversions to f16 (W_comb zero-padded 48 -> 64 cols)
  cvt(W_qkv, Wqkv_h, (size_t)DIM_ * QKVN_);
  cvt(kW1, kW1_h, (size_t)CDIM_ * CDIM_);
  cvt(vW1, vW1_h, (size_t)CDIM_ * CDIM_);
  cvt(kW2, kW2_h, (size_t)CDIM_ * DH_);
  cvt(vW2, vW2_h, (size_t)CDIM_ * DH_);
  cvt(W_out, Wout_h, (size_t)DIM_ * DIM_);
  padcvt_kernel<<<dim3((DIM_ * GATESN_ + 255) / 256), 256, 0, stream>>>(
      W_comb, Wcomb_h, DIM_, 48, GATESN_);
  padbias_kernel<<<1, GATESN_, 0, stream>>>(b_comb, bcomb_p, 48);

  // 3. QKV projection
  gemm_wmma_kernel<<<dim3(S_LEN / 64, QKVN_ / 64), 256, 0, stream>>>(
      xn_f16, Wqkv_h, nullptr, qkv_f32, nullptr, S_LEN, QKVN_, DIM_, 0, 0);

  // 4. RoPE + reshape + MLP inputs + qbar
  prep_kernel<<<S_LEN, 64, 0, stream>>>(qkv_f32, k_pos, v_pos, q_h, rq_h, rk_h,
                                        v_h, qbar, kcm, vcm);

  // 5. K/V compression MLPs
  gemm_wmma_kernel<<<dim3(1, CDIM_ / 64), 256, 0, stream>>>(
      kcm, kW1_h, kb1, nullptr, hk_h, 64, CDIM_, CDIM_, 1, 1);
  gemm_wmma_kernel<<<dim3(1, 1), 256, 0, stream>>>(
      hk_h, kW2_h, kb2, ckb_f32, nullptr, 64, DH_, CDIM_, 0, 0);
  gemm_wmma_kernel<<<dim3(1, CDIM_ / 64), 256, 0, stream>>>(
      vcm, vW1_h, vb1, nullptr, hv_h, 64, CDIM_, CDIM_, 1, 1);
  gemm_wmma_kernel<<<dim3(1, 1), 256, 0, stream>>>(
      hv_h, vW2_h, vb2, cvb_f32, nullptr, 64, DH_, CDIM_, 0, 0);

  // 6. assemble padded compressed K/V (mem token + 32 blocks + zero pad)
  build_ckcv_kernel<<<dim3(48, KVH_), 64, 0, stream>>>(ckb_f32, cvb_f32, mem_kv, ck_h, cv_h);

  // 7. compressed attention
  cattn_kernel<<<dim3(S_LEN / 16, HEADS_), 32, 0, stream>>>(q_h, ck_h, cv_h, c_out);

  // 8. block selection (top-8)
  select_kernel<<<dim3(S_LEN / 64, KVH_), 64, 0, stream>>>(qbar, ckb_f32, sel_idx, sel_ok);

  // 9. fine attention over selected + diagonal blocks
  fattn_kernel<<<dim3(S_LEN, KVH_), 32, 0, stream>>>(rq_h, rk_h, v_h, sel_idx, sel_ok, f_out);

  // 10. sliding-window attention
  swin_kernel<<<dim3(S_LEN / 16, HEADS_), 32, 0, stream>>>(rq_h, rk_h, v_h, s_out);

  // 11. gates (sigmoid fused, padded N=64)
  gemm_wmma_kernel<<<dim3(S_LEN / 64, 1), 256, 0, stream>>>(
      xn_f16, Wcomb_h, bcomb_p, gates, nullptr, S_LEN, GATESN_, DIM_, 2, 0);

  // 12. combine
  combine_kernel<<<dim3((S_LEN * DIM_) / 256), 256, 0, stream>>>(gates, c_out, f_out, s_out, comb_h);

  // 13. output projection -> d_out (f32)
  gemm_wmma_kernel<<<dim3(S_LEN / 64, DIM_ / 64), 256, 0, stream>>>(
      comb_h, Wout_h, nullptr, out, nullptr, S_LEN, DIM_, DIM_, 0, 0);
}
